// LM_32573031973132
// MI455X (gfx1250) — compile-verified, matrix operand order checked by reading, not run on hardware
//
#include <hip/hip_runtime.h>

// ---------------------------------------------------------------------------
// Problem constants (from the reference)
// ---------------------------------------------------------------------------
namespace cfg {
constexpr int VOCAB = 32000;
constexpr int H     = 1024;
constexpr int KD    = 512;
constexpr int VD    = 1024;
constexpr int OD    = 1024;
constexpr int LL    = 2;
constexpr int BB    = 4;
constexpr int TT    = 1024;
constexpr int MT    = BB * TT;   // 4096 rows in every GEMM
}

// ---------------------------------------------------------------------------
// CDNA5 WMMA types (wave32, 16x16x32 bf16 -> f32)
// ---------------------------------------------------------------------------
typedef __attribute__((ext_vector_type(16))) __bf16 v16bf;
typedef __attribute__((ext_vector_type(8)))  float  v8f;

__device__ __forceinline__ unsigned short f32_to_bf16(float f) {
    unsigned int u = __float_as_uint(f);
    u += 0x7FFFu + ((u >> 16) & 1u);      // round-to-nearest-even
    return (unsigned short)(u >> 16);
}

// ---------------------------------------------------------------------------
// Embedding gather + convert to bf16
// ---------------------------------------------------------------------------
__global__ __launch_bounds__(256) void embed_kernel(
    const int* __restrict__ ids, const float* __restrict__ emb,
    unsigned short* __restrict__ hA)
{
    const int row = blockIdx.x;                  // 0 .. B*T-1
    const int id  = ids[row];
    const float* src = emb + (size_t)id * cfg::H;
    unsigned short* dst = hA + (size_t)row * cfg::H;
    for (int i = threadIdx.x; i < cfg::H; i += blockDim.x)
        dst[i] = f32_to_bf16(src[i]);
}

// ---------------------------------------------------------------------------
// Weight convert + transpose:  W[k][n] f32  ->  Wt[n][k] bf16
// Tiled 32x32 through LDS (padded to kill bank conflicts); both global
// access patterns fully coalesced. Weights end up in the WMMA-friendly
// B layout so the GEMM stages them with pure b128 traffic.
// ---------------------------------------------------------------------------
__global__ __launch_bounds__(256) void cvtT_kernel(
    const float* __restrict__ src, unsigned short* __restrict__ dst,
    int Kdim, int Ndim)
{
    __shared__ float tile[32][33];
    const int kb = blockIdx.y * 32;
    const int nb = blockIdx.x * 32;
    const int tx = threadIdx.x & 31;
    const int ty = threadIdx.x >> 5;            // 0..7
#pragma unroll
    for (int j = 0; j < 32; j += 8)
        tile[ty + j][tx] = src[(size_t)(kb + ty + j) * Ndim + nb + tx];
    __syncthreads();
#pragma unroll
    for (int j = 0; j < 32; j += 8)
        dst[(size_t)(nb + ty + j) * Kdim + kb + tx] = f32_to_bf16(tile[tx][ty + j]);
}

// ---------------------------------------------------------------------------
// bf16 WMMA GEMM:  C[M,N] = act(A[M,K] @ Bt^T + bias),  Bt is N x K (bf16)
//   block tile 128x128, k-step 64 (2 WMMA k-steps), 256 threads = 8 waves
//   (2 x 4); each wave: 4x2 16x16 f32 accumulators -> 16 static
//   v_wmma_f32_16x16x32_bf16 per iteration, one barrier per iteration.
//   Double-buffered LDS (2 x 32KB). Staged registers are stored to LDS
//   immediately after the global loads (writing buf cur^1 while all waves
//   read buf cur is race-free), so no staging register is live across the
//   WMMA section -> fits the 128-VGPR budget with no scratch spills.
//   Latency of the blocking staging loads is covered by (a) WGP-scope
//   prefetch issued two tiles ahead and (b) the other 7 waves' WMMA work.
// ---------------------------------------------------------------------------
#define BM 128
#define BN 128
#define BKK 64

__global__ __launch_bounds__(256) void gemm_bf16_kernel(
    const unsigned short* __restrict__ A,   // M x K, bf16 row-major
    const unsigned short* __restrict__ Bt,  // N x K, bf16 row-major (transposed weights)
    const float* __restrict__ bias,         // N (may be null)
    float* __restrict__ Cf,                 // optional f32 out (M x N)
    unsigned short* __restrict__ Cb,        // optional bf16 out (M x N)
    int M, int N, int K, int act)           // act: 0=none, 1=sigmoid
{
    __shared__ alignas(32) unsigned short As[2][BM][BKK];  // 2 x 16 KB
    __shared__ alignas(32) unsigned short Bs[2][BN][BKK];  // 2 x 16 KB, [n][k]

    const int tid   = threadIdx.x;
    const int lane  = tid & 31;
    const int wave  = tid >> 5;          // 0..7
    const int wm    = wave & 1;          // 2 waves along M
    const int wn    = wave >> 1;         // 4 waves along N
    const int l16   = lane & 15;
    const int lhalf = lane >> 4;         // 0: K 0..15, 1: K 16..31
    const int bm0   = blockIdx.y * BM;
    const int bn0   = blockIdx.x * BN;

    // staging coordinates (identical pattern for A and Bt)
    const int sr = tid >> 1;             // row 0..127, 32 elems/thread
    const int sc = (tid & 1) * 32;

    const unsigned short* ap = A  + (size_t)(bm0 + sr) * K + sc;
    const unsigned short* bp = Bt + (size_t)(bn0 + sr) * K + sc;

    v8f acc[4][2] = {};

    // load tile and store it to LDS immediately (short register live range)
    auto stage_tile = [&](int buf) {
        uint4 a_reg[4], b_reg[4];
        const uint4* pa = (const uint4*)ap;
#pragma unroll
        for (int j = 0; j < 4; ++j) a_reg[j] = pa[j];
        const uint4* pb = (const uint4*)bp;
#pragma unroll
        for (int j = 0; j < 4; ++j) b_reg[j] = pb[j];
#pragma unroll
        for (int j = 0; j < 4; ++j)
            *(uint4*)&As[buf][sr][sc + j * 8] = a_reg[j];
#pragma unroll
        for (int j = 0; j < 4; ++j)
            *(uint4*)&Bs[buf][sr][sc + j * 8] = b_reg[j];
        ap += BKK;
        bp += BKK;
    };

    // prologue: stage tile 0
    stage_tile(0);
    __syncthreads();

    int cur = 0;
    for (int k0 = 0; k0 < K; k0 += BKK) {
        if (k0 + 2 * BKK < K) {                   // prefetch tile i+2,
            __builtin_prefetch(ap + BKK, 0, 3);   // high locality -> WGP scope
            __builtin_prefetch(bp + BKK, 0, 3);
        }
        if (k0 + BKK < K)
            stage_tile(cur ^ 1);                  // tile i+1 (prefetched earlier)

#pragma unroll
        for (int ks = 0; ks < 2; ++ks) {          // two 32-wide k-steps
#pragma unroll
            for (int mi = 0; mi < 4; ++mi) {
                const int row = wm * 64 + mi * 16 + l16;
                v16bf a = *reinterpret_cast<const v16bf*>(
                    &As[cur][row][ks * 32 + lhalf * 16]);
#pragma unroll
                for (int ni = 0; ni < 2; ++ni) {
                    const int col = wn * 32 + ni * 16 + l16;
                    v16bf b = *reinterpret_cast<const v16bf*>(
                        &Bs[cur][col][ks * 32 + lhalf * 16]);
                    acc[mi][ni] = __builtin_amdgcn_wmma_f32_16x16x32_bf16(
                        false, a, false, b, (short)0, acc[mi][ni], false, false);
                }
            }
        }

        __syncthreads();
        cur ^= 1;
    }

    // ---- epilogue: bias + optional sigmoid; f32 and/or bf16 stores ------
    // Lanes 0-15 and 16-31 of each store cover two contiguous 64B row
    // segments -> well-coalesced.
#pragma unroll
    for (int mi = 0; mi < 4; ++mi) {
#pragma unroll
        for (int ni = 0; ni < 2; ++ni) {
            const int col = bn0 + wn * 32 + ni * 16 + l16;
            const float bv = bias ? bias[col] : 0.0f;
#pragma unroll
            for (int i = 0; i < 8; ++i) {
                const int row = bm0 + wm * 64 + mi * 16 + (lhalf ? (8 + i) : i);
                float c = acc[mi][ni][i] + bv;
                if (act == 1) c = 1.0f / (1.0f + __expf(-c));
                const size_t idx = (size_t)row * N + col;
                if (Cf) Cf[idx] = c;
                if (Cb) Cb[idx] = f32_to_bf16(c);
            }
        }
    }
}

// ---------------------------------------------------------------------------
// Gated linear-attention scan, split-K x2, 4 timesteps per barrier pair.
//   state[k,v]_t = state[k,v]_{t-1} * g_t[k] + k_t[k] * v_t[v]
//   partial_t[v] = sum_{k in half} q_t[k] * state[k,v]_t
// Grid: (Vd/32, B, 2). 256 threads = 8 k-groups x 32 v-lanes; each thread
// keeps 32 k x 1 v of state in registers. f32 partials per k-half are
// summed by a tiny combine kernel (bandwidth-trivial).
// ---------------------------------------------------------------------------
__global__ __launch_bounds__(256) void scan_kernel(
    const float* __restrict__ q, const float* __restrict__ kk,
    const float* __restrict__ g, const float* __restrict__ v,
    float* __restrict__ pf)          // [2][B][T][VD] f32 partials
{
    const int vc  = blockIdx.x;          // v chunk of 32
    const int b   = blockIdx.y;
    const int kh  = blockIdx.z;          // k half (0/1)
    const int tid = threadIdx.x;
    const int vi  = tid & 31;
    const int kg  = tid >> 5;            // 0..7 (32 k each)
    const int vg  = vc * 32 + vi;
    const int kofs = kh * 256;

    float s[32];
#pragma unroll
    for (int i = 0; i < 32; ++i) s[i] = 0.0f;

    __shared__ float qsh[4][256];
    __shared__ float ksh[4][256];
    __shared__ float gsh[4][256];
    __shared__ float partial[4][8][32];

    float* out = pf + ((size_t)(kh * cfg::BB + b)) * cfg::TT * cfg::VD;
    const int l64 = tid & 63;

    for (int t0 = 0; t0 < cfg::TT; t0 += 4) {
        // ---- stage q/k/g for 4 timesteps (this k-half), float4 loads ----
#pragma unroll
        for (int pass = 0; pass < 3; ++pass) {
            const int combo = (tid >> 6) + pass * 4;      // 0..11
            const int tt  = combo / 3;                    // timestep 0..3
            const int arr = combo % 3;                    // q / k / g
            const size_t base =
                ((size_t)b * cfg::TT + t0 + tt) * cfg::KD + kofs + l64 * 4;
            const float* srcp = (arr == 0) ? q : (arr == 1) ? kk : g;
            const float4 val = *(const float4*)(srcp + base);
            if (arr == 0)      *(float4*)&qsh[tt][l64 * 4] = val;
            else if (arr == 1) *(float4*)&ksh[tt][l64 * 4] = val;
            else               *(float4*)&gsh[tt][l64 * 4] = val;
        }
        float vt[4];
#pragma unroll
        for (int tt = 0; tt < 4; ++tt)
            vt[tt] = v[((size_t)b * cfg::TT + t0 + tt) * cfg::VD + vg];
        __syncthreads();

        // ---- 4 recurrence steps, no barrier between them ----------------
        const int kbase = kg * 32;
#pragma unroll
        for (int tt = 0; tt < 4; ++tt) {
            float p = 0.0f;
#pragma unroll
            for (int i = 0; i < 32; ++i) {
                const float sn = s[i] * gsh[tt][kbase + i]
                               + ksh[tt][kbase + i] * vt[tt];
                s[i] = sn;
                p += qsh[tt][kbase + i] * sn;
            }
            partial[tt][kg][vi] = p;
        }
        __syncthreads();

        // ---- batched cross-wave reduction: kg 0..3 each reduce one tt ---
        if (kg < 4) {
            const int tt = kg;
            float sum = 0.0f;
#pragma unroll
            for (int w = 0; w < 8; ++w) sum += partial[tt][w][vi];
            out[(size_t)(t0 + tt) * cfg::VD + vg] = sum;
        }
        // staging for the next group only touches qsh/ksh/gsh (fully
        // consumed before the barrier above); partial[] is rewritten only
        // after the next post-stage barrier -> 2 barriers per 4 steps.
    }
}

// ---------------------------------------------------------------------------
// Combine the two split-K partials -> bf16 scan output
// ---------------------------------------------------------------------------
__global__ __launch_bounds__(256) void combine_kernel(
    const float* __restrict__ pf, unsigned short* __restrict__ oB, long n)
{
    long i      = (long)blockIdx.x * blockDim.x + threadIdx.x;
    long stride = (long)gridDim.x * blockDim.x;
    for (; i < n; i += stride)
        oB[i] = f32_to_bf16(pf[i] + pf[i + n]);
}

// ---------------------------------------------------------------------------
// Host-side orchestration
// ---------------------------------------------------------------------------
static void launch_gemm(const unsigned short* A, const unsigned short* Bt,
                        const float* bias, float* Cf, unsigned short* Cb,
                        int M, int N, int K, int act, hipStream_t st)
{
    dim3 grid(N / BN, M / BM), block(256);
    gemm_bf16_kernel<<<grid, block, 0, st>>>(A, Bt, bias, Cf, Cb, M, N, K, act);
}

static void launch_cvtT(const float* src, unsigned short* dst,
                        int Kdim, int Ndim, hipStream_t st)
{
    dim3 grid(Ndim / 32, Kdim / 32), block(256);
    cvtT_kernel<<<grid, block, 0, st>>>(src, dst, Kdim, Ndim);
}

extern "C" void kernel_launch(void* const* d_in, const int* in_sizes, int n_in,
                              void* d_out, int out_size, void* d_ws, size_t ws_size,
                              hipStream_t stream)
{
    using namespace cfg;
    (void)in_sizes; (void)n_in; (void)out_size; (void)ws_size;

    const int*   ids    = (const int*)  d_in[0];
    const float* emb    = (const float*)d_in[1];
    const float* Wq     = (const float*)d_in[2];
    const float* bq     = (const float*)d_in[3];
    const float* Wk     = (const float*)d_in[4];
    const float* bk     = (const float*)d_in[5];
    const float* Wg     = (const float*)d_in[6];
    const float* bg     = (const float*)d_in[7];
    const float* Wv     = (const float*)d_in[8];
    const float* bv     = (const float*)d_in[9];
    const float* Wo     = (const float*)d_in[10];
    const float* bo     = (const float*)d_in[11];
    const float* W_head = (const float*)d_in[12];
    const float* b_head = (const float*)d_in[13];
    float* logits       = (float*)d_out;

    // ---- workspace layout (deterministic, 256B aligned) -----------------
    unsigned char* w = (unsigned char*)d_ws;
    size_t off = 0;
    auto take = [&](size_t bytes) -> unsigned char* {
        unsigned char* p = w + off;
        off = (off + bytes + 255) & ~(size_t)255;
        return p;
    };
    unsigned short* hA  = (unsigned short*)take((size_t)MT * H  * 2);  // current h (bf16)
    unsigned short* oB  = (unsigned short*)take((size_t)MT * VD * 2);  // scan output (bf16)
    float*          qf  = (float*)take((size_t)MT * KD * 4);
    float*          kf  = (float*)take((size_t)MT * KD * 4);
    float*          gf  = (float*)take((size_t)MT * KD * 4);
    float*          vf  = (float*)take((size_t)MT * VD * 4);
    float*          pf  = (float*)take((size_t)2 * MT * VD * 4);       // split-K partials
    unsigned short* WqT = (unsigned short*)take((size_t)LL * H  * KD * 2);  // [l][KD][H]
    unsigned short* WkT = (unsigned short*)take((size_t)LL * H  * KD * 2);
    unsigned short* WgT = (unsigned short*)take((size_t)LL * H  * KD * 2);
    unsigned short* WvT = (unsigned short*)take((size_t)LL * H  * VD * 2);  // [l][VD][H]
    unsigned short* WoT = (unsigned short*)take((size_t)LL * VD * OD * 2);  // [l][OD][VD]
    unsigned short* WhT = (unsigned short*)take((size_t)OD * VOCAB * 2);    // [VOCAB][OD]

    // ---- 1) embedding gather -> bf16 ------------------------------------
    embed_kernel<<<dim3(MT), dim3(256), 0, stream>>>(ids, emb, hA);

    // ---- 2) weight conversions f32 -> bf16, transposed to [n][k] --------
    for (int l = 0; l < LL; ++l) {
        launch_cvtT(Wq + (size_t)l * H * KD,  WqT + (size_t)l * H * KD,  H,  KD, stream);
        launch_cvtT(Wk + (size_t)l * H * KD,  WkT + (size_t)l * H * KD,  H,  KD, stream);
        launch_cvtT(Wg + (size_t)l * H * KD,  WgT + (size_t)l * H * KD,  H,  KD, stream);
        launch_cvtT(Wv + (size_t)l * H * VD,  WvT + (size_t)l * H * VD,  H,  VD, stream);
        launch_cvtT(Wo + (size_t)l * VD * OD, WoT + (size_t)l * VD * OD, VD, OD, stream);
    }
    launch_cvtT(W_head, WhT, OD, VOCAB, stream);

    // ---- 3) per-layer pipeline ------------------------------------------
    for (int l = 0; l < LL; ++l) {
        const unsigned short* wq = WqT + (size_t)l * H * KD;
        const unsigned short* wk = WkT + (size_t)l * H * KD;
        const unsigned short* wg = WgT + (size_t)l * H * KD;
        const unsigned short* wv = WvT + (size_t)l * H * VD;
        const unsigned short* wo = WoT + (size_t)l * VD * OD;

        launch_gemm(hA, wq, bq + (size_t)l * KD, qf, nullptr, MT, KD, H, 0, stream);
        launch_gemm(hA, wk, bk + (size_t)l * KD, kf, nullptr, MT, KD, H, 1, stream);
        launch_gemm(hA, wg, bg + (size_t)l * KD, gf, nullptr, MT, KD, H, 1, stream);
        launch_gemm(hA, wv, bv + (size_t)l * VD, vf, nullptr, MT, VD, H, 0, stream);

        scan_kernel<<<dim3(VD / 32, BB, 2), dim3(256), 0, stream>>>(qf, kf, gf, vf, pf);
        combine_kernel<<<dim3(8192), dim3(256), 0, stream>>>(pf, oB, (long)MT * VD);

        // h_{l+1} = o @ Wo + bo  (bf16 output feeds next layer / head)
        launch_gemm(oB, wo, bo + (size_t)l * OD, nullptr, hA, MT, OD, VD, 0, stream);
    }

    // ---- 4) head GEMM: logits = h @ W_head + b_head (f32 out) -----------
    launch_gemm(hA, WhT, b_head, logits, nullptr, MT, VOCAB, OD, 0, stream);
}
